// Net_63874753626763
// MI455X (gfx1250) — matrix operand order repaired and checked by reading, not compiled, not run on hardware
//
#include <hip/hip_runtime.h>
#include <stdint.h>

typedef __attribute__((ext_vector_type(16))) _Float16       v16h;
typedef __attribute__((ext_vector_type(8)))  float          v8f;
typedef __attribute__((ext_vector_type(8)))  unsigned short v8us;

#define BETA 0.9375f
#define THR  1.0f

__device__ __forceinline__ uint32_t hash32(uint32_t x) {
    x *= 0x9E3779B1u;
    x ^= x >> 16; x *= 0x85EBCA6Bu;
    x ^= x >> 13; x *= 0xC2B2AE35u;
    x ^= x >> 16;
    return x;
}

// One workgroup = 16 batch rows, 8 wave32 waves. Wave w owns hidden units
// [16w,16w+16). Entire T-step recurrence runs in-kernel: membrane state in
// registers (WMMA C-layout), weights in LDS as pre-swizzled f16 B-fragments,
// input spikes regenerated on-chip from a hash RNG. Double-buffered spike /
// hidden-spike fragments -> ONE barrier per timestep, so gen-VALU of step
// t+1 co-executes with WMMA of step t across the two waves per SIMD.
__global__ __launch_bounds__(256) void snn_scan_kernel(
    const float* __restrict__ data,  // [B,784]
    const float* __restrict__ w1,    // [128,784]
    const float* __restrict__ b1,    // [128]
    const float* __restrict__ w2,    // [10,128]
    const float* __restrict__ b2,    // [10]
    const int*   __restrict__ nsp,   // [1] num_steps
    float*       __restrict__ out,   // [T,B,10] spk2 ++ [T,B,10] mem2
    int B)
{
    // ---- LDS: 294,400 B (needs CDNA5's 320KB WGP LDS) ----
    __shared__ __align__(32) _Float16 w1s[8][25][32][16];    // layer1 B-frags, 200KB
    __shared__ __align__(32) _Float16 spkA[2][25][32][16];   // input-spike A-frags, 2x25.6KB
    __shared__ __align__(32) _Float16 w2s[4][32][16];        // layer2 B-frags (N pad 10->16), 4KB
    __shared__ __align__(32) _Float16 spk1A[2][4][32][16];   // hidden-spike A-frags, 2x4KB
    __shared__ __align__(16) unsigned short thr16[16][800];  // Bernoulli thresholds, 25.6KB
    __shared__ float b1s[128];

    const int tid     = threadIdx.x;
    const int wave    = tid >> 5;
    const int lane    = tid & 31;
    const int laneN   = lane & 15;   // column / row-in-tile selector
    const int laneHi  = lane >> 4;
    const int rowBase = blockIdx.x * 16;
    const int T       = nsp[0];

    // ---- one-time staging ----
    for (int idx = tid; idx < 8 * 25 * 32 * 16; idx += 256) {
        int i  = idx & 15;
        int L  = (idx >> 4) & 31;
        int c  = (idx >> 9) % 25;
        int tN = (idx >> 9) / 25;
        int n  = tN * 16 + (L & 15);
        int k  = c * 32 + ((i >> 3) * 16) + ((L >> 4) * 8) + (i & 7);
        w1s[tN][c][L][i] = (_Float16)((k < 784) ? w1[n * 784 + k] : 0.0f);
    }
    for (int idx = tid; idx < 4 * 32 * 16; idx += 256) {
        int i = idx & 15;
        int L = (idx >> 4) & 31;
        int c = idx >> 9;
        int n = L & 15;
        int k = c * 32 + ((i >> 3) * 16) + ((L >> 4) * 8) + (i & 7);
        w2s[c][L][i] = (_Float16)((n < 10) ? w2[n * 128 + k] : 0.0f);
    }
    for (int idx = tid; idx < 16 * 800; idx += 256) {
        int row = idx / 800;
        int k   = idx % 800;
        float p = (k < 784) ? data[(rowBase + row) * 784 + k] : 0.0f;
        uint32_t u = (uint32_t)(p * 65536.0f);
        if (u > 65535u) u = 65535u;
        thr16[row][k] = (unsigned short)u;
    }
    if (tid < 128) b1s[tid] = b1[tid];

    // generate input spikes for timestep ts into buffer bsel, A-frag layout:
    // lane L: row = L&15; half i<8  -> K = 32c + 8*laneHi + i
    //                     half i>=8 -> K = 32c + 16 + 8*laneHi + (i-8)
    auto gen_spikes = [&](int ts, int bsel) {
        const uint32_t rowg = (uint32_t)(rowBase + laneN);
        for (int c = wave; c < 25; c += 8) {
            const int kb = c * 32 + laneHi * 8;
            const v8us tlo = *(const v8us*)&thr16[laneN][kb];
            const v8us thi = *(const v8us*)&thr16[laneN][kb + 16];
            v16h av;
#pragma unroll
            for (int i = 0; i < 8; ++i) {
                uint32_t h = hash32(((uint32_t)ts * 2048u + rowg) * 800u
                                    + (uint32_t)(kb + i));
                av[i] = ((h >> 16) < (uint32_t)tlo[i]) ? (_Float16)1.0f : (_Float16)0.0f;
            }
#pragma unroll
            for (int i = 0; i < 8; ++i) {
                uint32_t h = hash32(((uint32_t)ts * 2048u + rowg) * 800u
                                    + (uint32_t)(kb + 16 + i));
                av[8 + i] = ((h >> 16) < (uint32_t)thi[i]) ? (_Float16)1.0f : (_Float16)0.0f;
            }
            *(v16h*)&spkA[bsel][c][lane][0] = av;
        }
    };

    __syncthreads();          // staging (thr16) complete
    gen_spikes(0, 0);
    __syncthreads();          // step-0 spikes visible

    // ---- persistent state (WMMA C-layout: VGPR r -> row r+8*laneHi, col laneN) ----
    v8f mem1 = {}; v8f spk1 = {};
    v8f mem2 = {}; v8f spk2 = {};
    const float b1r = b1s[wave * 16 + laneN];
    float b2r = 0.0f;
    if (wave == 0 && laneN < 10) b2r = b2[laneN];
    const long long memOff = (long long)T * B * 10;

    // per-lane constant destination pattern for writing this wave's hidden
    // spikes directly into layer-2 A-fragment layout:
    // hidden unit h = 16*wave + laneN -> chunk c2 = wave>>1, ko = 16*(wave&1)+laneN
    const int c2    = wave >> 1;
    const int ko    = 16 * (wave & 1) + laneN;
    const int i2    = (ko & 7) + ((ko >= 16) ? 8 : 0);
    const int lOff2 = ((ko >> 3) & 1) * 16;

    for (int t = 0; t < T; ++t) {
        const int cur = t & 1;

        // spikes for t+1 (independent buffer: co-executes with WMMA below)
        gen_spikes(t + 1, cur ^ 1);

        // --- layer 1: 25 chained WMMAs per wave ---
        v8f acc = {};
#pragma unroll
        for (int c = 0; c < 25; ++c) {
            v16h a = *(const v16h*)&spkA[cur][c][lane][0];
            v16h b = *(const v16h*)&w1s[wave][c][lane][0];
            acc = __builtin_amdgcn_wmma_f32_16x16x32_f16(
                false, a, false, b, (short)0, acc, false, false);
        }
        // LIF update; hidden spikes stored directly as layer-2 A-fragments
#pragma unroll
        for (int r = 0; r < 8; ++r) {
            float m = BETA * mem1[r] + acc[r] + b1r - spk1[r] * THR;
            float s = (m - THR > 0.0f) ? 1.0f : 0.0f;
            mem1[r] = m; spk1[r] = s;
            int row = r + laneHi * 8;
            spk1A[cur][c2][row + lOff2][i2] = (_Float16)s;
        }

        __syncthreads();   // the ONLY barrier per timestep

        // --- layer 2: wave 0 only; overlaps other waves' next-step work ---
        if (wave == 0) {
            v8f acc2 = {};
#pragma unroll
            for (int c = 0; c < 4; ++c) {
                v16h a = *(const v16h*)&spk1A[cur][c][lane][0];
                v16h b = *(const v16h*)&w2s[c][lane][0];
                acc2 = __builtin_amdgcn_wmma_f32_16x16x32_f16(
                    false, a, false, b, (short)0, acc2, false, false);
            }
#pragma unroll
            for (int r = 0; r < 8; ++r) {
                float m = BETA * mem2[r] + acc2[r] + b2r - spk2[r] * THR;
                float s = (m - THR > 0.0f) ? 1.0f : 0.0f;
                mem2[r] = m; spk2[r] = s;
                if (laneN < 10) {
                    int rowg = rowBase + r + laneHi * 8;
                    long long o = ((long long)t * B + rowg) * 10 + laneN;
                    out[o] = s;            // spk2_rec
                    out[memOff + o] = m;   // mem2_rec
                }
            }
        }
    }
}

extern "C" void kernel_launch(void* const* d_in, const int* in_sizes, int n_in,
                              void* d_out, int out_size, void* d_ws, size_t ws_size,
                              hipStream_t stream) {
    const float* data = (const float*)d_in[0];  // [B,784]
    const float* w1   = (const float*)d_in[1];  // [128,784]
    const float* b1   = (const float*)d_in[2];  // [128]
    const float* w2   = (const float*)d_in[3];  // [10,128]
    const float* b2   = (const float*)d_in[4];  // [10]
    const int*   ns   = (const int*)d_in[5];    // scalar num_steps

    const int B = in_sizes[0] / 784;            // 2048
    const int blocks = B / 16;                  // 128 WGs x 8 waves

    snn_scan_kernel<<<blocks, 256, 0, stream>>>(data, w1, b1, w2, b2, ns,
                                                (float*)d_out, B);
}
